// TACF_7009386627630
// MI455X (gfx1250) — compile-verified, matrix operand order checked by reading, not run on hardware
//
#include <hip/hip_runtime.h>
#include <hip/hip_bf16.h>

typedef __attribute__((ext_vector_type(16))) _Float16 v16h;
typedef __attribute__((ext_vector_type(8)))  float    v8f;

#define NREAL 2516
#define NPAD  2528
#define NTILM 158
#define CDIM  128
#define QKVD  384
#define MLPD  512

// load halfs p[0..7] and p[16..23] (two b128)
__device__ inline v16h ldfrag(const _Float16* p) {
  v16h r;
  uint4* rp = reinterpret_cast<uint4*>(&r);
  rp[0] = *reinterpret_cast<const uint4*>(p);
  rp[1] = *reinterpret_cast<const uint4*>(p + 16);
  return r;
}
// load 8 contiguous halfs into K=0..15 region, upper K zeroed (dh=16 fragments)
__device__ inline v16h ldfrag8(const _Float16* p) {
  v16h r;
  uint4* rp = reinterpret_cast<uint4*>(&r);
  rp[0] = *reinterpret_cast<const uint4*>(p);
  rp[1] = uint4{0u, 0u, 0u, 0u};
  return r;
}

// ---------------- pack x = [task_descriptor ; support(NHWC)] , zero pads ----
__global__ void pack_x_kernel(const float* __restrict__ td,
                              const float* __restrict__ sup,
                              float* __restrict__ x)
{
  int idx = blockIdx.x * blockDim.x + threadIdx.x;
  if (idx >= NPAD * CDIM) return;
  int c = idx & 127;
  int t = idx >> 7;
  float v = 0.f;
  if (t < 16) v = td[t * CDIM + c];
  else if (t < NREAL) {
    int s = t - 16;
    int b = s / 100, p = s % 100;
    v = sup[(size_t)b * (CDIM * 100) + c * 100 + p];
  }
  x[idx] = v;
}

// ---------------- weight transpose to f16:  WT[n*K+k] = W[k*N+n] ------------
__global__ void wtransp_kernel(const float* __restrict__ W,
                               _Float16* __restrict__ WT, int K, int N)
{
  int idx = blockIdx.x * blockDim.x + threadIdx.x;
  if (idx >= K * N) return;
  int n = idx % N, k = idx / N;
  WT[(size_t)n * K + k] = (_Float16)W[(size_t)k * N + n];
}

// ---------------- LayerNorm (last dim = 128) -> f16 rows, zero pad rows -----
__global__ void ln_rows_kernel(const float* __restrict__ X,
                               const float* __restrict__ g,
                               const float* __restrict__ b,
                               _Float16* __restrict__ H16, int Mreal)
{
  int row = blockIdx.x;
  int lane = threadIdx.x;                      // 32 threads
  _Float16* hr = H16 + (size_t)row * CDIM;
  if (row >= Mreal) {
    #pragma unroll
    for (int i = 0; i < 4; ++i) hr[lane + 32 * i] = (_Float16)0.f;
    return;
  }
  const float* xr = X + (size_t)row * CDIM;
  float v[4]; float s = 0.f;
  #pragma unroll
  for (int i = 0; i < 4; ++i) { v[i] = xr[lane + 32 * i]; s += v[i]; }
  #pragma unroll
  for (int off = 16; off >= 1; off >>= 1) s += __shfl_xor(s, off, 32);
  float mean = s * (1.f / 128.f);
  float q = 0.f;
  #pragma unroll
  for (int i = 0; i < 4; ++i) { float d = v[i] - mean; q += d * d; }
  #pragma unroll
  for (int off = 16; off >= 1; off >>= 1) q += __shfl_xor(q, off, 32);
  float rstd = rsqrtf(q * (1.f / 128.f) + 1e-5f);
  #pragma unroll
  for (int i = 0; i < 4; ++i) {
    int c = lane + 32 * i;
    hr[c] = (_Float16)((v[i] - mean) * rstd * g[c] + b[c]);
  }
}

// ---------------- WMMA GEMM: each wave computes a 16(M) x 64(N) tile --------
// A: Mpad x K f16 row-major.  BT: N x K f16 row-major (i.e. B transposed).
// Register double-buffered: fragments for k+32 are in flight while the four
// WMMAs for k execute, so the s_wait before the WMMAs covers year-old loads.
__global__ __launch_bounds__(128)
void gemm_wmma_kernel(const _Float16* __restrict__ A,
                      const _Float16* __restrict__ BT,
                      const float* __restrict__ bias,
                      const float* __restrict__ Res,
                      float* __restrict__ C32,
                      _Float16* __restrict__ C16,
                      int Mreal, int N, int K, int act)
{
  const int lane = threadIdx.x & 31;
  const int wave = threadIdx.x >> 5;
  const int tileM = blockIdx.y * 4 + wave;      // waves split along M
  if (tileM >= NTILM) return;                   // wave-uniform exit
  const int tileN = blockIdx.x;                 // 64 output columns per wave
  const int g   = lane >> 4;
  const int l15 = lane & 15;

  const _Float16* ap = A  + (size_t)(tileM * 16 + l15) * K + g * 8;
  const _Float16* bp = BT + (size_t)(tileN * 64 + l15) * K + g * 8;
  const size_t bstep = (size_t)16 * K;          // halfs between 16-col subtiles

  v8f acc0 = {}, acc1 = {}, acc2 = {}, acc3 = {};
  v16h a  = ldfrag(ap);
  v16h b0 = ldfrag(bp);
  v16h b1 = ldfrag(bp + bstep);
  v16h b2 = ldfrag(bp + 2 * bstep);
  v16h b3 = ldfrag(bp + 3 * bstep);

  int kb = 0;
  for (; kb + 32 < K; kb += 32) {
    v16h an  = ldfrag(ap + kb + 32);
    v16h bn0 = ldfrag(bp + kb + 32);
    v16h bn1 = ldfrag(bp + bstep + kb + 32);
    v16h bn2 = ldfrag(bp + 2 * bstep + kb + 32);
    v16h bn3 = ldfrag(bp + 3 * bstep + kb + 32);
    acc0 = __builtin_amdgcn_wmma_f32_16x16x32_f16(false, a, false, b0, (short)0, acc0, false, false);
    acc1 = __builtin_amdgcn_wmma_f32_16x16x32_f16(false, a, false, b1, (short)0, acc1, false, false);
    acc2 = __builtin_amdgcn_wmma_f32_16x16x32_f16(false, a, false, b2, (short)0, acc2, false, false);
    acc3 = __builtin_amdgcn_wmma_f32_16x16x32_f16(false, a, false, b3, (short)0, acc3, false, false);
    a = an; b0 = bn0; b1 = bn1; b2 = bn2; b3 = bn3;
  }
  acc0 = __builtin_amdgcn_wmma_f32_16x16x32_f16(false, a, false, b0, (short)0, acc0, false, false);
  acc1 = __builtin_amdgcn_wmma_f32_16x16x32_f16(false, a, false, b1, (short)0, acc1, false, false);
  acc2 = __builtin_amdgcn_wmma_f32_16x16x32_f16(false, a, false, b2, (short)0, acc2, false, false);
  acc3 = __builtin_amdgcn_wmma_f32_16x16x32_f16(false, a, false, b3, (short)0, acc3, false, false);

  v8f accs[4] = {acc0, acc1, acc2, acc3};
  #pragma unroll
  for (int sub = 0; sub < 4; ++sub) {
    const int n = tileN * 64 + sub * 16 + l15;
    const float bv = bias ? bias[n] : 0.f;
    #pragma unroll
    for (int e = 0; e < 8; ++e) {
      int m = tileM * 16 + g * 8 + e;           // C layout: vgpr e, half g
      if (m < Mreal) {
        float v = accs[sub][e] + bv;
        if (Res) v += Res[(size_t)m * N + n];
        if (act == 1) v = 0.5f * v * (1.f + erff(v * 0.70710678f));  // exact GELU
        if (C32) C32[(size_t)m * N + n] = v;
        if (C16) C16[(size_t)m * N + n] = (_Float16)v;
      } else {
        if (C16) C16[(size_t)m * N + n] = (_Float16)0.f;             // zero pads
      }
    }
  }
}

// ---------------- pack per-head f16 Q/K/V for b128 fragment loads -----------
// Q16,K16: [head][token][16] (Q pre-scaled by dh^-0.5); VT16: [head][dh][token]
__global__ void pack_qkv_kernel(const float* __restrict__ qkv,
                                _Float16* __restrict__ Q16,
                                _Float16* __restrict__ K16,
                                _Float16* __restrict__ VT16)
{
  int idx = blockIdx.x * blockDim.x + threadIdx.x;
  if (idx >= 8 * NPAD * 16) return;
  int d  = idx & 15;
  int t  = (idx >> 4) % NPAD;
  int hd = idx / (16 * NPAD);
  float q = 0.f, k = 0.f, v = 0.f;
  if (t < NREAL) {
    const float* row = qkv + (size_t)t * QKVD + hd * 16 + d;
    q = row[0] * 0.25f;
    k = row[CDIM];
    v = row[2 * CDIM];
  }
  Q16[idx] = (_Float16)q;
  K16[idx] = (_Float16)k;
  VT16[((size_t)hd * 16 + d) * NPAD + t] = (_Float16)v;
}

// ---------------- flash attention: wave per (head, 16-row query tile) -------
// dh = 16 -> one 16x16x32 WMMA (upper 16 K zero) per score / PV tile.
__global__ __launch_bounds__(32)
void attn_flash_kernel(const _Float16* __restrict__ Q16,
                       const _Float16* __restrict__ K16,
                       const _Float16* __restrict__ VT16,
                       _Float16* __restrict__ O16)
{
  __shared__ __align__(16) _Float16 p_s[256];
  const int lane = threadIdx.x & 31;
  const int g    = lane >> 4;
  const int l15  = lane & 15;
  const int qt = blockIdx.x;            // 0..157
  const int hd = blockIdx.y;            // 0..7

  const v16h aq = ldfrag8(Q16 + ((size_t)hd * NPAD + qt * 16 + l15) * 16 + g * 8);
  const _Float16* kbase = K16 + (size_t)hd * NPAD * 16 + l15 * 16 + g * 8;
  const _Float16* vbase = VT16 + ((size_t)hd * 16 + l15) * NPAD + g * 8;

  v8f acc = {};
  float Mrow[8], Lrow[8];
  #pragma unroll
  for (int e = 0; e < 8; ++e) { Mrow[e] = -3.0e38f; Lrow[e] = 0.f; }

  for (int kt = 0; kt < NTILM; ++kt) {
    const v16h bk = ldfrag8(kbase + (size_t)kt * 256);       // 16 tokens * 16 halfs
    v8f z = {};
    v8f s = __builtin_amdgcn_wmma_f32_16x16x32_f16(false, aq, false, bk,
                                                   (short)0, z, false, false);
    const bool kvalid = (kt * 16 + l15) < NREAL;
    #pragma unroll
    for (int e = 0; e < 8; ++e) {                            // online softmax/row
      float sv = kvalid ? s[e] : -3.0e38f;
      float mx = sv;
      #pragma unroll
      for (int off = 8; off >= 1; off >>= 1) mx = fmaxf(mx, __shfl_xor(mx, off, 32));
      float mnew  = fmaxf(Mrow[e], mx);
      float alpha = __expf(Mrow[e] - mnew);
      float p     = __expf(sv - mnew);
      float ps = p;
      #pragma unroll
      for (int off = 8; off >= 1; off >>= 1) ps += __shfl_xor(ps, off, 32);
      Lrow[e] = Lrow[e] * alpha + ps;
      acc[e] *= alpha;
      Mrow[e] = mnew;
      p_s[(g * 8 + e) * 16 + l15] = (_Float16)p;             // C-layout -> LDS
    }
    __syncthreads();
    const v16h ap = ldfrag8(p_s + l15 * 16 + g * 8);         // A-layout readback
    const v16h bv = ldfrag8(vbase + kt * 16);                // V^T: 8 tokens/lane
    acc = __builtin_amdgcn_wmma_f32_16x16x32_f16(false, ap, false, bv,
                                                 (short)0, acc, false, false);
    __syncthreads();
  }

  #pragma unroll
  for (int e = 0; e < 8; ++e) {
    int m = qt * 16 + g * 8 + e;
    float o = (m < NREAL && Lrow[e] > 0.f) ? acc[e] / Lrow[e] : 0.f;
    O16[(size_t)m * CDIM + hd * 16 + l15] = (_Float16)o;
  }
}

// ---------------- region: sigmoid(feat . key0 / 128) ------------------------
__global__ void region_kernel(const float* __restrict__ feat,
                              const float* __restrict__ key0,  // x rows 0..15
                              float* __restrict__ reg, int B)
{
  int idx = blockIdx.x * blockDim.x + threadIdx.x;
  if (idx >= B * 16 * 100) return;
  int p = idx % 100, m = (idx / 100) % 16, b = idx / 1600;
  const float* fp = feat + (size_t)b * (CDIM * 100) + p;
  const float* kp = key0 + m * CDIM;
  float d = 0.f;
  for (int c = 0; c < CDIM; ++c) d += fp[c * 100] * kp[c];
  reg[idx] = 1.f / (1.f + __expf(-d * (1.f / 128.f)));
}

// ---------------- fuse: feat * (mean_m(region) + 1) -------------------------
__global__ void fuse_kernel(const float* __restrict__ feat,
                            const float* __restrict__ reg,
                            float* __restrict__ fused, int B)
{
  int idx = blockIdx.x * blockDim.x + threadIdx.x;
  if (idx >= B * CDIM * 100) return;
  int p = idx % 100, b = idx / (CDIM * 100);
  float mm = 0.f;
  for (int m = 0; m < 16; ++m) mm += reg[(size_t)b * 1600 + m * 100 + p];
  fused[idx] = feat[idx] * (mm * (1.f / 16.f) + 1.f);
}

// ---------------- channel LayerNorm (over C, eps 1e-6) ----------------------
__global__ void ln_chan_kernel(const float* __restrict__ in,
                               const float* __restrict__ g,
                               const float* __restrict__ b,
                               float* __restrict__ out, int B)
{
  int idx = blockIdx.x * blockDim.x + threadIdx.x;
  if (idx >= B * 100) return;
  int p = idx % 100, bb = idx / 100;
  const float* ip = in + (size_t)bb * (CDIM * 100) + p;
  float s = 0.f;
  for (int c = 0; c < CDIM; ++c) s += ip[c * 100];
  float mean = s * (1.f / 128.f);
  float q = 0.f;
  for (int c = 0; c < CDIM; ++c) { float d = ip[c * 100] - mean; q += d * d; }
  float rstd = rsqrtf(q * (1.f / 128.f) + 1e-6f);
  float* op = out + (size_t)bb * (CDIM * 100) + p;
  for (int c = 0; c < CDIM; ++c)
    op[c * 100] = (ip[c * 100] - mean) * rstd * g[c] + b[c];
}

// ---------------- 3x3 conv, pad 1, act: 0 none / 1 relu / 2 sigmoid ---------
__global__ void conv3x3_kernel(const float* __restrict__ in,
                               const float* __restrict__ w,
                               float* __restrict__ out,
                               int B, int Cin, int Cout, int act)
{
  int idx = blockIdx.x * blockDim.x + threadIdx.x;
  if (idx >= B * Cout * 100) return;
  int p = idx % 100, oc = (idx / 100) % Cout, b = idx / (100 * Cout);
  int iy = p / 10, ix = p % 10;
  float acc = 0.f;
  for (int ic = 0; ic < Cin; ++ic) {
    const float* ip = in + ((size_t)b * Cin + ic) * 100;
    const float* wp = w + ((size_t)oc * Cin + ic) * 9;
    #pragma unroll
    for (int ky = 0; ky < 3; ++ky) {
      int y = iy + ky - 1; if ((unsigned)y >= 10u) continue;
      #pragma unroll
      for (int kx = 0; kx < 3; ++kx) {
        int xx = ix + kx - 1; if ((unsigned)xx >= 10u) continue;
        acc += ip[y * 10 + xx] * wp[ky * 3 + kx];
      }
    }
  }
  if (act == 1) acc = fmaxf(acc, 0.f);
  else if (act == 2) acc = 1.f / (1.f + __expf(-acc));
  out[idx] = acc;
}

// ---------------- out[b,c,t] = sum_p sig[b,t,p]*fused[b,c,p] / 100 ----------
__global__ void rfm_out_kernel(const float* __restrict__ sig,
                               const float* __restrict__ fused,
                               float* __restrict__ out, int B)
{
  int idx = blockIdx.x * blockDim.x + threadIdx.x;
  if (idx >= B * CDIM * 32) return;
  int t = idx % 32, c = (idx / 32) % CDIM, b = idx / (32 * CDIM);
  const float* sp = sig + ((size_t)b * 32 + t) * 100;
  const float* fp = fused + ((size_t)b * CDIM + c) * 100;
  float acc = 0.f;
  for (int p = 0; p < 100; ++p) acc += sp[p] * fp[p];
  out[idx] = acc * 0.01f;
}

// ---------------------------------------------------------------------------
static inline int cdiv(int a, int b) { return (a + b - 1) / b; }

extern "C" void kernel_launch(void* const* d_in, const int* in_sizes, int n_in,
                              void* d_out, int out_size, void* d_ws, size_t ws_size,
                              hipStream_t stream)
{
  (void)in_sizes; (void)n_in; (void)out_size; (void)ws_size;
  const float* sup   = (const float*)d_in[0];
  const float* qry   = (const float*)d_in[1];
  const float* td    = (const float*)d_in[2];
  const float* ln1g  = (const float*)d_in[3];
  const float* ln1b  = (const float*)d_in[4];
  const float* qkvw  = (const float*)d_in[5];
  const float* qkvb  = (const float*)d_in[6];
  const float* outw  = (const float*)d_in[7];
  const float* outb  = (const float*)d_in[8];
  const float* ln2g  = (const float*)d_in[9];
  const float* ln2b  = (const float*)d_in[10];
  const float* w1    = (const float*)d_in[11];
  const float* b1    = (const float*)d_in[12];
  const float* w2    = (const float*)d_in[13];
  const float* b2    = (const float*)d_in[14];
  const float* cg    = (const float*)d_in[15];
  const float* cb    = (const float*)d_in[16];
  const float* cw1   = (const float*)d_in[17];
  const float* cw2   = (const float*)d_in[18];
  const float* cw3   = (const float*)d_in[19];
  const float* cw4   = (const float*)d_in[20];

  char* ws = (char*)d_ws;
  // workspace layout (all offsets 16B-aligned); total ~23.3 MB
  float*     x    = (float*)    (ws + 0);            // 2528x128 f32
  _Float16*  h16  = (_Float16*) (ws + 1294336);      // 2528x128 f16
  float*     qkv  = (float*)    (ws + 1941504);      // 2528x384 f32
  _Float16*  o16  = (_Float16*) (ws + 5824512);      // 2528x128 f16
  _Float16*  t16  = (_Float16*) (ws + 6471680);      // 2528x512 f16
  char*      wtb  =              ws + 9060352;       // f16 transposed weights
  float*     fusS = (float*)    (ws + 9846784);      // 25x128x100
  float*     fusQ = (float*)    (ws + 11126784);     // 75x128x100
  float*     regS = (float*)    (ws + 14966784);     // 25x16x100
  float*     regQ = (float*)    (ws + 15126784);     // 75x16x100
  float*     lnb  = (float*)    (ws + 15606784);     // <=75x128x100
  float*     cA   = (float*)    (ws + 19446784);     // <=75x32x100
  float*     cB   = (float*)    (ws + 20406784);     // <=75x32x100
  _Float16*  Q16  = (_Float16*) (ws + 21366784);     // 8x2528x16
  _Float16*  K16  = (_Float16*) (ws + 22013952);     // 8x2528x16
  _Float16*  VT16 = (_Float16*) (ws + 22661120);     // 8x16x2528

  const size_t LSTRIDE = 393216;                     // per-layer f16 weight bytes
  auto qkvT = [&](int l){ return (_Float16*)(wtb + l * LSTRIDE + 0); };
  auto outT = [&](int l){ return (_Float16*)(wtb + l * LSTRIDE + 98304); };
  auto w1T  = [&](int l){ return (_Float16*)(wtb + l * LSTRIDE + 131072); };
  auto w2T  = [&](int l){ return (_Float16*)(wtb + l * LSTRIDE + 262144); };

  const int TPB = 256;
  const int GY = cdiv(NTILM, 4);                     // 4 M-tiles (waves) per block

  pack_x_kernel<<<cdiv(NPAD * CDIM, TPB), TPB, 0, stream>>>(td, sup, x);

  for (int l = 0; l < 2; ++l) {
    wtransp_kernel<<<cdiv(CDIM * QKVD, TPB), TPB, 0, stream>>>(qkvw + (size_t)l * CDIM * QKVD, qkvT(l), CDIM, QKVD);
    wtransp_kernel<<<cdiv(CDIM * CDIM, TPB), TPB, 0, stream>>>(outw + (size_t)l * CDIM * CDIM, outT(l), CDIM, CDIM);
    wtransp_kernel<<<cdiv(CDIM * MLPD, TPB), TPB, 0, stream>>>(w1 + (size_t)l * CDIM * MLPD, w1T(l), CDIM, MLPD);
    wtransp_kernel<<<cdiv(MLPD * CDIM, TPB), TPB, 0, stream>>>(w2 + (size_t)l * MLPD * CDIM, w2T(l), MLPD, CDIM);
  }

  for (int l = 0; l < 2; ++l) {
    // h = LN1(x) -> f16
    ln_rows_kernel<<<NPAD, 32, 0, stream>>>(x, ln1g + l * CDIM, ln1b + l * CDIM, h16, NREAL);
    // qkv = h @ qkv_w + qkv_b
    gemm_wmma_kernel<<<dim3(QKVD / 64, GY), 128, 0, stream>>>(
        h16, qkvT(l), qkvb + l * QKVD, nullptr, qkv, nullptr, NREAL, QKVD, CDIM, 0);
    // per-head f16 packing for single-b128 fragment loads
    pack_qkv_kernel<<<cdiv(8 * NPAD * 16, TPB), TPB, 0, stream>>>(qkv, Q16, K16, VT16);
    // flash attention -> o16 (f16, pads zeroed)
    attn_flash_kernel<<<dim3(NTILM, 8), 32, 0, stream>>>(Q16, K16, VT16, o16);
    // x = o @ out_w + out_b + x
    gemm_wmma_kernel<<<dim3(CDIM / 64, GY), 128, 0, stream>>>(
        o16, outT(l), outb + l * CDIM, x, x, nullptr, NREAL, CDIM, CDIM, 0);
    // h = LN2(x) -> f16
    ln_rows_kernel<<<NPAD, 32, 0, stream>>>(x, ln2g + l * CDIM, ln2b + l * CDIM, h16, NREAL);
    // t16 = gelu(h @ w1 + b1)  (f16 only)
    gemm_wmma_kernel<<<dim3(MLPD / 64, GY), 128, 0, stream>>>(
        h16, w1T(l), b1 + l * MLPD, nullptr, nullptr, t16, NREAL, MLPD, CDIM, 1);
    // x = t @ w2 + b2 + x
    gemm_wmma_kernel<<<dim3(CDIM / 64, GY), 128, 0, stream>>>(
        t16, w2T(l), b2 + l * CDIM, x, x, nullptr, NREAL, CDIM, MLPD, 0);
  }

  // key0 = x[0:16,:] (read in-place). Region maps + fuse.
  region_kernel<<<cdiv(75 * 16 * 100, TPB), TPB, 0, stream>>>(qry, x, regQ, 75);
  region_kernel<<<cdiv(25 * 16 * 100, TPB), TPB, 0, stream>>>(sup, x, regS, 25);
  fuse_kernel<<<cdiv(25 * CDIM * 100, TPB), TPB, 0, stream>>>(sup, regS, fusS, 25);
  fuse_kernel<<<cdiv(75 * CDIM * 100, TPB), TPB, 0, stream>>>(qry, regQ, fusQ, 75);

  float* out = (float*)d_out;
  // rfm(asp): support set (B=25) -> out[0 : 25*128*32)
  {
    const int B = 25;
    ln_chan_kernel<<<cdiv(B * 100, TPB), TPB, 0, stream>>>(fusS, cg, cb, lnb, B);
    conv3x3_kernel<<<cdiv(B * 32 * 100, TPB), TPB, 0, stream>>>(lnb, cw1, cA, B, CDIM, 32, 1);
    conv3x3_kernel<<<cdiv(B * 32 * 100, TPB), TPB, 0, stream>>>(cA, cw2, cB, B, 32, 32, 1);
    conv3x3_kernel<<<cdiv(B * 32 * 100, TPB), TPB, 0, stream>>>(cB, cw3, cA, B, 32, 32, 1);
    conv3x3_kernel<<<cdiv(B * 32 * 100, TPB), TPB, 0, stream>>>(cA, cw4, cB, B, 32, 32, 2);
    rfm_out_kernel<<<cdiv(B * CDIM * 32, TPB), TPB, 0, stream>>>(cB, fusS, out, B);
  }
  // rfm(aq): query set (B=75) -> out[25*128*32 : )
  {
    const int B = 75;
    ln_chan_kernel<<<cdiv(B * 100, TPB), TPB, 0, stream>>>(fusQ, cg, cb, lnb, B);
    conv3x3_kernel<<<cdiv(B * 32 * 100, TPB), TPB, 0, stream>>>(lnb, cw1, cA, B, CDIM, 32, 1);
    conv3x3_kernel<<<cdiv(B * 32 * 100, TPB), TPB, 0, stream>>>(cA, cw2, cB, B, 32, 32, 1);
    conv3x3_kernel<<<cdiv(B * 32 * 100, TPB), TPB, 0, stream>>>(cB, cw3, cA, B, 32, 32, 1);
    conv3x3_kernel<<<cdiv(B * 32 * 100, TPB), TPB, 0, stream>>>(cA, cw4, cB, B, 32, 32, 2);
    rfm_out_kernel<<<cdiv(B * CDIM * 32, TPB), TPB, 0, stream>>>(cB, fusQ, out + 25 * CDIM * 32, B);
  }
}
